// Net_3642132267012
// MI455X (gfx1250) — matrix operand order repaired and checked by reading, not compiled, not run on hardware
//
#include <hip/hip_runtime.h>

// ---------------------------------------------------------------------------
// GNN forward for MI455X (gfx1250, wave32).
//  - activations stored bf16 between layers (halves edge-gather traffic,
//    feeds V_WMMA_F32_16X16X32_BF16 directly)
//  - scatter: f32 atomics into L2-resident accumulator
//  - GEMM: fused agg@W_rel + b + x@W_root, bias+ReLU; x tile DMA'd to LDS
//    via GLOBAL_LOAD_ASYNC_TO_LDS_B128 when the toolchain exposes it
// ---------------------------------------------------------------------------

typedef __attribute__((ext_vector_type(16))) __bf16        v16bf;
typedef __attribute__((ext_vector_type(8)))  float         v8f;
typedef __attribute__((ext_vector_type(8)))  unsigned int  v8u;
typedef __attribute__((ext_vector_type(4)))  int           v4i;

#define N_NODES 51200
#define N_EDGES 819200
#define HD      128
#define NG      128
#define SEG     400
#define NOUT    10

#if __has_builtin(__builtin_amdgcn_global_load_async_to_lds_b128)
#define HAVE_ASYNC_LDS 1
typedef __attribute__((address_space(1))) v4i* gv4i_p;   // global int4*
typedef __attribute__((address_space(3))) v4i* lv4i_p;   // LDS int4*
#else
#define HAVE_ASYNC_LDS 0
#endif

__device__ __forceinline__ unsigned short f2bf(float f) {
    unsigned int u = __builtin_bit_cast(unsigned int, f);
    u += 0x7FFFu + ((u >> 16) & 1u);            // round-to-nearest-even
    return (unsigned short)(u >> 16);
}
__device__ __forceinline__ float bf2f(unsigned short h) {
    return __builtin_bit_cast(float, (unsigned int)h << 16);
}
__device__ __forceinline__ unsigned int pack2bf(float a, float b) {
    return (unsigned int)f2bf(a) | ((unsigned int)f2bf(b) << 16);
}

// ---------------------------------------------------------------------------
__global__ void zero_f4(float4* __restrict__ p, int n4) {
    int i = blockIdx.x * blockDim.x + threadIdx.x;
    if (i < n4) p[i] = make_float4(0.f, 0.f, 0.f, 0.f);
}

// fp32 [N,HD] -> bf16 [N,HD]   (1 thread = 4 elems)
__global__ void cvt_x_bf16(const float* __restrict__ x, unsigned short* __restrict__ y) {
    int i = blockIdx.x * blockDim.x + threadIdx.x;        // N*HD/4 threads
    float4 v = *(const float4*)(x + (size_t)i * 4);
    uint2 o;
    o.x = pack2bf(v.x, v.y);
    o.y = pack2bf(v.z, v.w);
    *(uint2*)(y + (size_t)i * 4) = o;
}

// 4 fp32 [k][n] weight matrices (128x128) -> transposed bf16 [n][k]
__global__ void cvt_weightsT(const float* __restrict__ w0, const float* __restrict__ w1,
                             const float* __restrict__ w2, const float* __restrict__ w3,
                             unsigned short* __restrict__ out) {
    int i = blockIdx.x * blockDim.x + threadIdx.x;        // 4 * 16384
    int m = i >> 14, r = i & 16383;
    int k = r >> 7,  n = r & 127;
    const float* src = (m == 0) ? w0 : (m == 1) ? w1 : (m == 2) ? w2 : w3;
    out[m * 16384 + n * HD + k] = f2bf(src[k * HD + n]);
}

// ---------------------------------------------------------------------------
// Edge scatter: agg[dst] += x[src] (x bf16 -> f32 atomics).  1 wave / edge,
// each lane moves 4 features (8B bf16 load, 4x global_atomic_add_f32).
__global__ void scatter_add(const unsigned short* __restrict__ x,
                            const long long* __restrict__ src,
                            const long long* __restrict__ dst,
                            float* __restrict__ agg) {
    int e = blockIdx.x * 8 + (threadIdx.x >> 5);
    if (e >= N_EDGES) return;
    int lane = threadIdx.x & 31;
    long long s = src[e];
    long long d = dst[e];
    int e2 = e + 8;
    if (e2 < N_EDGES)                     // global_prefetch hint for next edge row
        __builtin_prefetch((const void*)(x + (size_t)src[e2] * HD + lane * 4), 0, 0);
    uint2 pv = *(const uint2*)(x + (size_t)s * HD + lane * 4);
    float* a = agg + (size_t)d * HD + lane * 4;
    atomicAdd(a + 0, bf2f((unsigned short)(pv.x & 0xFFFFu)));
    atomicAdd(a + 1, bf2f((unsigned short)(pv.x >> 16)));
    atomicAdd(a + 2, bf2f((unsigned short)(pv.y & 0xFFFFu)));
    atomicAdd(a + 3, bf2f((unsigned short)(pv.y >> 16)));
}

// ---------------------------------------------------------------------------
// Fused GraphConv GEMM: t = relu(agg @ W_rel + b + x @ W_root)
// Block = 16 node rows x 128 cols, 8 waves; wave owns a 16x16 tile.
// agg tile: fp32 -> bf16 staged through VALU; x tile: bf16 DMA'd LDS-direct.
// LDS row pitch 136 halfwords (272B) keeps b128 chunks 16B-aligned.
__global__ void __launch_bounds__(256, 2)
gemm_conv(const float* __restrict__ agg, const unsigned short* __restrict__ xbf,
          const unsigned short* __restrict__ wrelT,
          const unsigned short* __restrict__ wrootT,
          const float* __restrict__ bias, float* __restrict__ out) {
    __shared__ unsigned short sA[16][136];   // agg tile (bf16)
    __shared__ unsigned short sX[16][136];   // x tile   (bf16)
    const int row0 = blockIdx.x * 16;

    // x tile: 16 rows x 256B = 256 lanes x 16B
    {
        int r   = threadIdx.x >> 4;          // 0..15
        int c8  = (threadIdx.x & 15) * 8;    // 8 halfwords = 16B
        const unsigned short* g = xbf + (size_t)(row0 + r) * HD + c8;
#if HAVE_ASYNC_LDS
        __builtin_amdgcn_global_load_async_to_lds_b128(
            (gv4i_p)(unsigned short*)g,
            (lv4i_p)&sX[r][c8],
            0, 0);
#else
        *(uint4*)&sX[r][c8] = *(const uint4*)g;
#endif
    }
    // agg tile: fp32 load + convert
    for (int i = threadIdx.x; i < 16 * HD; i += 256) {
        int r = i >> 7, c = i & 127;
        sA[r][c] = f2bf(agg[(size_t)(row0 + r) * HD + c]);
    }
#if HAVE_ASYNC_LDS
#if __has_builtin(__builtin_amdgcn_s_wait_asynccnt)
    __builtin_amdgcn_s_wait_asynccnt(0);
#else
    asm volatile("s_wait_asynccnt 0" ::: "memory");
#endif
#endif
    __syncthreads();

    const int lane = threadIdx.x & 31;
    const int wave = threadIdx.x >> 5;
    const int col0 = wave * 16;
    const int m    = lane & 15;
    const int half = lane >> 4;

    v8f acc = {0.f, 0.f, 0.f, 0.f, 0.f, 0.f, 0.f, 0.f};

#pragma unroll
    for (int ks = 0; ks < 4; ++ks) {
        const int kb = ks * 32;
        v8u ar, ax, br, bx;
#pragma unroll
        for (int v = 0; v < 8; ++v) {
            // A 16x32 bf16: VGPR v = K pair; lanes 0-15 K{0..7,16..23},
            // lanes 16-31 K{8..15,24..31}
            int ka = kb + ((v >> 2) << 4) + (half << 3) + ((v & 3) << 1);
            ar[v] = *(const unsigned int*)&sA[m][ka];
            ax[v] = *(const unsigned int*)&sX[m][ka];
            // B 32x16 bf16: same K order as A; lanes 0-15 K{0..7,16..23},
            // lanes 16-31 K{8..15,24..31}; VGPR v = K pair in transposed weights
            int kbK = kb + ((v >> 2) << 4) + (half << 3) + ((v & 3) << 1);
            br[v] = *(const unsigned int*)&wrelT [(size_t)(col0 + m) * HD + kbK];
            bx[v] = *(const unsigned int*)&wrootT[(size_t)(col0 + m) * HD + kbK];
        }
        acc = __builtin_amdgcn_wmma_f32_16x16x32_bf16(
                  false, __builtin_bit_cast(v16bf, ar),
                  false, __builtin_bit_cast(v16bf, br),
                  (short)0, acc, false, false);
        acc = __builtin_amdgcn_wmma_f32_16x16x32_bf16(
                  false, __builtin_bit_cast(v16bf, ax),
                  false, __builtin_bit_cast(v16bf, bx),
                  (short)0, acc, false, false);
    }

    const float bn = bias[col0 + m];
#pragma unroll
    for (int r = 0; r < 8; ++r) {
        float val = acc[r] + bn;                     // D: M = r + 8*half, N = m
        val = val > 0.f ? val : 0.f;
        out[(size_t)(row0 + r + 8 * half) * HD + col0 + m] = val;
    }
}

// ---------------------------------------------------------------------------
// Mixup + bf16 store: y = bf16(lam*t + (1-lam)*t[perm])
__global__ void mixup_k(const float* __restrict__ t,
                        const long long* __restrict__ perm,
                        const float* __restrict__ lam,
                        unsigned short* __restrict__ y) {
    int i = blockIdx.x * blockDim.x + threadIdx.x;
    int node = i >> 5;
    if (node >= N_NODES) return;
    int c = (i & 31) * 4;
    float l  = lam[0];
    float il = 1.f - l;
    long long p = perm[node];
    float4 a = *(const float4*)(t + (size_t)node * HD + c);
    float4 b = *(const float4*)(t + (size_t)p    * HD + c);
    uint2 o;
    o.x = pack2bf(l * a.x + il * b.x, l * a.y + il * b.y);
    o.y = pack2bf(l * a.z + il * b.z, l * a.w + il * b.w);
    *(uint2*)(y + (size_t)node * HD + c) = o;
}

// ---------------------------------------------------------------------------
__global__ void pool_k(const unsigned short* __restrict__ x, float* __restrict__ pooled) {
    int g = blockIdx.x;
    int c = threadIdx.x;
    const unsigned short* p = x + (size_t)g * SEG * HD + c;
    float s = 0.f;
    for (int i = 0; i < SEG; ++i) s += bf2f(p[(size_t)i * HD]);
    pooled[g * HD + c] = s;
}

__global__ void head_k(const float* __restrict__ pooled,
                       const float* __restrict__ W, const float* __restrict__ b,
                       float* __restrict__ out) {
    int g = blockIdx.x * blockDim.x + threadIdx.x;
    if (g >= NG) return;
    float logits[NOUT];
#pragma unroll
    for (int o = 0; o < NOUT; ++o) logits[o] = b[o];
    for (int k = 0; k < HD; ++k) {
        float pv = pooled[g * HD + k];
#pragma unroll
        for (int o = 0; o < NOUT; ++o) logits[o] += pv * W[k * NOUT + o];
    }
    float mx = logits[0];
#pragma unroll
    for (int o = 1; o < NOUT; ++o) mx = fmaxf(mx, logits[o]);
    float se = 0.f;
#pragma unroll
    for (int o = 0; o < NOUT; ++o) se += __expf(logits[o] - mx);
    float ls = __logf(se);
#pragma unroll
    for (int o = 0; o < NOUT; ++o) out[g * NOUT + o] = (logits[o] - mx) - ls;
}

// ---------------------------------------------------------------------------
extern "C" void kernel_launch(void* const* d_in, const int* in_sizes, int n_in,
                              void* d_out, int out_size, void* d_ws, size_t ws_size,
                              hipStream_t stream) {
    const float*     x0     = (const float*)d_in[0];
    const long long* ei     = (const long long*)d_in[1];   // [2, E] int64
    const float*     lam    = (const float*)d_in[2];
    // d_in[3]=ptr, d_in[4]=batch: implied by contiguous segment layout
    const long long* perm1  = (const long long*)d_in[5];
    const long long* perm2  = (const long long*)d_in[6];
    const long long* perm3  = (const long long*)d_in[7];
    const float* W1_rel  = (const float*)d_in[8];
    const float* b1_rel  = (const float*)d_in[9];
    const float* W1_root = (const float*)d_in[10];
    const float* W2_rel  = (const float*)d_in[11];
    const float* b2_rel  = (const float*)d_in[12];
    const float* W2_root = (const float*)d_in[13];
    const float* W_lin   = (const float*)d_in[14];
    const float* b_lin   = (const float*)d_in[15];
    float* out = (float*)d_out;

    const long long* src = ei;
    const long long* dst = ei + N_EDGES;

    // workspace layout
    const size_t NB = (size_t)N_NODES * HD * sizeof(float);           // 26.2 MB
    const size_t HB = (size_t)N_NODES * HD * sizeof(unsigned short);  // 13.1 MB
    char* ws = (char*)d_ws;
    float*          agg    = (float*)(ws);                    // fp32 accumulator
    float*          t      = (float*)(ws + NB);               // fp32 gemm output
    unsigned short* xA     = (unsigned short*)(ws + 2 * NB);
    unsigned short* xB     = (unsigned short*)(ws + 2 * NB + HB);
    unsigned short* x0b    = (unsigned short*)(ws + 2 * NB + 2 * HB);
    unsigned short* wbf    = (unsigned short*)(ws + 2 * NB + 3 * HB);
    float*          pooled = (float*)(ws + 2 * NB + 3 * HB + 4 * 16384 * sizeof(unsigned short));

    const int n4       = N_NODES * HD / 4;       // 1,638,400
    const int zeroGrid = n4 / 256;               // 6400
    const int cvtGrid  = n4 / 256;               // 6400
    const int scatGrid = N_EDGES / 8;            // 102400
    const int gemmGrid = N_NODES / 16;           // 3200
    const int mixGrid  = (N_NODES * 32) / 256;   // 6400

    cvt_weightsT<<<(4 * 16384) / 256, 256, 0, stream>>>(W1_rel, W1_root, W2_rel, W2_root, wbf);
    cvt_x_bf16  <<<cvtGrid, 256, 0, stream>>>(x0, x0b);

    // ---- layer 1: x0b -> xA
    zero_f4    <<<zeroGrid, 256, 0, stream>>>((float4*)agg, n4);
    scatter_add<<<scatGrid, 256, 0, stream>>>(x0b, src, dst, agg);
    gemm_conv  <<<gemmGrid, 256, 0, stream>>>(agg, x0b, wbf + 0 * 16384, wbf + 1 * 16384, b1_rel, t);
    mixup_k    <<<mixGrid, 256, 0, stream>>>(t, perm1, lam, xA);

    // ---- layer 2: xA -> xB
    zero_f4    <<<zeroGrid, 256, 0, stream>>>((float4*)agg, n4);
    scatter_add<<<scatGrid, 256, 0, stream>>>(xA, src, dst, agg);
    gemm_conv  <<<gemmGrid, 256, 0, stream>>>(agg, xA, wbf + 2 * 16384, wbf + 3 * 16384, b2_rel, t);
    mixup_k    <<<mixGrid, 256, 0, stream>>>(t, perm2, lam, xB);

    // ---- layer 3: xB -> xA (conv2 weights reused, per reference)
    zero_f4    <<<zeroGrid, 256, 0, stream>>>((float4*)agg, n4);
    scatter_add<<<scatGrid, 256, 0, stream>>>(xB, src, dst, agg);
    gemm_conv  <<<gemmGrid, 256, 0, stream>>>(agg, xB, wbf + 2 * 16384, wbf + 3 * 16384, b2_rel, t);
    mixup_k    <<<mixGrid, 256, 0, stream>>>(t, perm3, lam, xA);

    // ---- pooling + head
    pool_k<<<NG, HD, 0, stream>>>(xA, pooled);
    head_k<<<1, NG, 0, stream>>>(pooled, W_lin, b_lin, out);
}
